// PerNeuronMLPHead_39367670235980
// MI455X (gfx1250) — matrix operand order repaired and checked by reading, not compiled, not run on hardware
//
#include <hip/hip_runtime.h>
#include <hip/hip_bf16.h>

typedef __attribute__((ext_vector_type(16))) _Float16 v16h;
typedef __attribute__((ext_vector_type(8)))  _Float16 v8h;
typedef __attribute__((ext_vector_type(8)))  float    v8f;
typedef __attribute__((ext_vector_type(4)))  int      v4i;

#define GLOBAL_AS __attribute__((address_space(1)))
#define LDS_AS    __attribute__((address_space(3)))

#if defined(__has_builtin)
#if __has_builtin(__builtin_amdgcn_global_load_async_to_lds_b128) && \
    __has_builtin(__builtin_amdgcn_s_wait_asynccnt)
#define USE_ASYNC_LDS 1
#endif
#endif

// Branchless tanh-form gelu:  x * sigmoid(1.59576912*(x + 0.044715*x^3)).
// Lowers to v_exp_f32 + v_rcp_f32 (TRANS ops co-execute with WMMA) + a few FMAs.
__device__ __forceinline__ float gelu_fast(float x) {
    const float x2 = x * x;
    const float t  = -1.5957691216f * x * __builtin_fmaf(0.044715f, x2, 1.0f);
    const float e  = __expf(t);
    return x * __builtin_amdgcn_rcpf(1.0f + e);
}

// ---------------------------------------------------------------------------
// Kernel 1: fold weights.  Wc = bp_w @ w1a (512x256), Wd = up_w @ w1b,
// bc = bp_b @ w1a, bd = up_b @ w1b + b1, and pre-swizzle w2 -> f16 WMMA
// B-fragment order:  chunk c = ((kk*8+nt)*2 + h)*32 + lane holds halfs
// e = h*8..h*8+7 of the 32x16 B tile (K = (e<8 ? e : e+8) + 8*(lane>>4), N = nt*16+lane&15).
// ---------------------------------------------------------------------------
__global__ __launch_bounds__(256) void fuse_weights_k(
    const float* __restrict__ bp_w, const float* __restrict__ bp_b,
    const float* __restrict__ up_w, const float* __restrict__ up_b,
    const float* __restrict__ w1,   const float* __restrict__ b1,
    const float* __restrict__ w2,
    float* __restrict__ Wc, float* __restrict__ Wd,
    float* __restrict__ bc, float* __restrict__ bd,
    _Float16* __restrict__ w2s)
{
    __shared__ float row[256];
    const int b = blockIdx.x, t = threadIdx.x;
    if (b < 512) {
        row[t] = bp_w[b * 256 + t];
        __syncthreads();
        float acc = 0.f;
        #pragma unroll 4
        for (int h = 0; h < 256; ++h) acc += row[h] * w1[h * 256 + t];
        Wc[b * 256 + t] = acc;
    } else if (b < 1024) {
        const int d = b - 512;
        row[t] = up_w[d * 256 + t];
        __syncthreads();
        float acc = 0.f;
        #pragma unroll 4
        for (int h = 0; h < 256; ++h) acc += row[h] * w1[(256 + h) * 256 + t];
        Wd[d * 256 + t] = acc;
    } else if (b == 1024) {
        float a0 = 0.f, a1 = 0.f;
        for (int h = 0; h < 256; ++h) {
            a0 += bp_b[h] * w1[h * 256 + t];
            a1 += up_b[h] * w1[(256 + h) * 256 + t];
        }
        bc[t] = a0;
        bd[t] = a1 + b1[t];
    } else {
        // w2 swizzle: 32768 halfs over blocks 1025..1152
        const int s    = (b - 1025) * 256 + t;
        const int j    = s & 7;
        const int c    = s >> 3;
        const int lane = c & 31;
        const int h    = (c >> 5) & 1;
        const int q    = c >> 6;
        const int nt   = q & 7;
        const int kk   = q >> 3;
        const int e    = h * 8 + j;
        const int k    = kk * 32 + 8 * (lane >> 4) + 16 * h + j;
        const int n    = nt * 16 + (lane & 15);
        w2s[s] = (_Float16)w2[k * 128 + n];
    }
}

// ---------------------------------------------------------------------------
// Kernel 2: Ah = f16(bin @ Wc + bc)  (1024x256),  Bh = f16(unit @ Wd + bd)
// ---------------------------------------------------------------------------
__global__ __launch_bounds__(256) void compute_ab_k(
    const float* __restrict__ bin, const float* __restrict__ unit,
    const float* __restrict__ Wc,  const float* __restrict__ Wd,
    const float* __restrict__ bc,  const float* __restrict__ bd,
    _Float16* __restrict__ Ah, _Float16* __restrict__ Bh)
{
    __shared__ float row[512];
    const int b = blockIdx.x, t = threadIdx.x;
    if (b < 1024) {
        row[t]       = bin[b * 512 + t];
        row[t + 256] = bin[b * 512 + 256 + t];
        __syncthreads();
        float acc = bc[t];
        #pragma unroll 4
        for (int d = 0; d < 512; ++d) acc += row[d] * Wc[d * 256 + t];
        Ah[b * 256 + t] = (_Float16)acc;
    } else {
        const int u = b - 1024;
        row[t]       = unit[u * 512 + t];
        row[t + 256] = unit[u * 512 + 256 + t];
        __syncthreads();
        float acc = bd[t];
        #pragma unroll 4
        for (int d = 0; d < 512; ++d) acc += row[d] * Wd[d * 256 + t];
        Bh[u * 256 + t] = (_Float16)acc;
    }
}

// ---------------------------------------------------------------------------
// Kernel 3: fused  gelu(A[i]+B[u]) @ w2  ->  gelu(+b2) @ w3  -> clip.
// Block = (i, u_half of 128 u's). 8 waves; wave w owns u-tile rows
// r = w*16 + (lane&15), all 128 n columns, K = 256.
// nt-outer / kk-inner with prefetched B-fragments: one live accumulator,
// ds loads for fragment (kk+1) issue before the WMMA consuming fragment kk.
// Prologue fill uses GLOBAL_LOAD_ASYNC_TO_LDS_B128 (ASYNCcnt) when available.
// ---------------------------------------------------------------------------
#define SBH_BYTES 67584   // 128 * 264 * 2
#define SW2_BYTES 65536   // 32768 halfs
#define SMEM_TOTAL (SBH_BYTES + SW2_BYTES + 512 + 512 + 512)  // 134656

__global__ __launch_bounds__(256) void pernueron_main_k(
    const _Float16* __restrict__ Ah, const _Float16* __restrict__ Bh,
    const _Float16* __restrict__ w2s,
    const float* __restrict__ b2, const float* __restrict__ w3,
    const float* __restrict__ b3p, float* __restrict__ out)
{
    extern __shared__ char smem[];
    _Float16* s_bh = (_Float16*)smem;                           // 128 x 264
    _Float16* s_w2 = (_Float16*)(smem + SBH_BYTES);             // 32768
    _Float16* s_a  = (_Float16*)(smem + SBH_BYTES + SW2_BYTES); // 256
    float*    s_b2 = (float*)(smem + SBH_BYTES + SW2_BYTES + 512);
    float*    s_w3 = (float*)(smem + SBH_BYTES + SW2_BYTES + 1024);

    const int i   = blockIdx.x >> 1;
    const int uh  = blockIdx.x & 1;
    const int tid = threadIdx.x;

#ifdef USE_ASYNC_LDS
    // ---- async fire-and-forget fill: global -> LDS, tracked by ASYNCcnt ----
    {
        GLOBAL_AS v4i* gB = (GLOBAL_AS v4i*)(Bh + uh * 128 * 256);
        LDS_AS v4i*    lB = (LDS_AS v4i*)s_bh;
        #pragma unroll
        for (int c = tid; c < 4096; c += 256) {
            const int rr = c >> 5, off = c & 31;
            __builtin_amdgcn_global_load_async_to_lds_b128(
                gB + c, lB + (rr * 33 + off), 0, 0);
        }
        GLOBAL_AS v4i* gW = (GLOBAL_AS v4i*)w2s;
        LDS_AS v4i*    lW = (LDS_AS v4i*)s_w2;
        #pragma unroll
        for (int c = tid; c < 4096; c += 256) {
            __builtin_amdgcn_global_load_async_to_lds_b128(gW + c, lW + c, 0, 0);
        }
        if (tid < 32) {
            GLOBAL_AS v4i* gA = (GLOBAL_AS v4i*)(Ah + i * 256);
            LDS_AS v4i*    lA = (LDS_AS v4i*)s_a;
            __builtin_amdgcn_global_load_async_to_lds_b128(gA + tid, lA + tid, 0, 0);
        }
        if (tid < 128) s_b2[tid] = b2[tid];
        else           s_w3[tid - 128] = w3[tid - 128];
        __builtin_amdgcn_s_wait_asynccnt(0);
    }
#else
    // ---- cooperative LDS fill (all 16B-aligned uint4 moves) ----
    {
        const uint4* src = (const uint4*)(Bh + uh * 128 * 256);
        uint4* dst = (uint4*)s_bh;
        #pragma unroll
        for (int c = tid; c < 4096; c += 256) {
            const int rr = c >> 5, off = c & 31;
            dst[rr * 33 + off] = src[c];         // 33 uint4 per padded row
        }
        const uint4* wsrc = (const uint4*)w2s;
        uint4* wdst = (uint4*)s_w2;
        #pragma unroll
        for (int c = tid; c < 4096; c += 256) wdst[c] = wsrc[c];
        if (tid < 32) ((uint4*)s_a)[tid] = ((const uint4*)(Ah + i * 256))[tid];
        if (tid < 128) s_b2[tid] = b2[tid];
        else           s_w3[tid - 128] = w3[tid - 128];
    }
#endif
    __syncthreads();

    const int wave = tid >> 5;
    const int lane = tid & 31;
    const int hi   = lane >> 4;
    const int lo   = lane & 15;
    const int r    = wave * 16 + lo;   // h-tile row (= local u), A-frag M = lane%16

    // ---- build all 8 A-fragments (gelu of A[i]+B[u]) once ----
    // A-fragment K map (16-bit A 16x32): K = (e<8 ? e : e+8) + 8*hi + 32*kk
    v16h af[8];
    #pragma unroll
    for (int kk = 0; kk < 8; ++kk) {
        const int kb = kk * 32 + hi * 8;
        const v8h a0 = *(const v8h*)(s_a + kb);
        const v8h a1 = *(const v8h*)(s_a + kb + 16);
        const v8h c0 = *(const v8h*)(s_bh + r * 264 + kb);
        const v8h c1 = *(const v8h*)(s_bh + r * 264 + kb + 16);
        #pragma unroll
        for (int e = 0; e < 8; ++e) {
            const float x0 = (float)a0[e] + (float)c0[e];
            const float x1 = (float)a1[e] + (float)c1[e];
            af[kk][e]     = (_Float16)gelu_fast(x0);
            af[kk][e + 8] = (_Float16)gelu_fast(x1);
        }
    }

    const float b3v = b3p[0];
    float partial[8];
    #pragma unroll
    for (int v = 0; v < 8; ++v) partial[v] = 0.f;

    const v8f vz = {0.f, 0.f, 0.f, 0.f, 0.f, 0.f, 0.f, 0.f};
    const _Float16* w2base = s_w2 + lane * 8;   // per-lane base; frag f at +f*512 halfs

    // prefetch fragment (kk=0, nt=0)
    v8h p0 = *(const v8h*)(w2base);
    v8h p1 = *(const v8h*)(w2base + 256);

    #pragma unroll
    for (int nt = 0; nt < 8; ++nt) {
        v8f acc = vz;
        #pragma unroll
        for (int kk = 0; kk < 8; ++kk) {
            v16h bf;
            #pragma unroll
            for (int e = 0; e < 8; ++e) { bf[e] = p0[e]; bf[e + 8] = p1[e]; }
            // prefetch next fragment: (kk+1, nt) or (0, nt+1)
            if (!(nt == 7 && kk == 7)) {
                const int nk = (kk == 7) ? 0 : kk + 1;
                const int nn = (kk == 7) ? nt + 1 : nt;
                const _Float16* np = w2base + (nk * 8 + nn) * 512;
                p0 = *(const v8h*)np;
                p1 = *(const v8h*)(np + 256);
            }
            acc = __builtin_amdgcn_wmma_f32_16x16x32_f16(
                false, af[kk], false, bf, (short)0, acc, false, false);
        }
        // ---- epilogue for this nt: y = gelu(acc + b2), partial += y*w3 ----
        // C/D layout: VGPR v -> row M = v + 8*hi, column N = nt*16 + lo.
        const int n = nt * 16 + lo;
        const float bb = s_b2[n], ww = s_w3[n];
        #pragma unroll
        for (int v = 0; v < 8; ++v) {
            partial[v] += gelu_fast(acc[v] + bb) * ww;
        }
    }

    // reduce over the 16 lanes of each half-wave (masks < 16 stay in-half)
    #pragma unroll
    for (int v = 0; v < 8; ++v) {
        float p = partial[v];
        p += __shfl_xor(p, 1, 32);
        p += __shfl_xor(p, 2, 32);
        p += __shfl_xor(p, 4, 32);
        p += __shfl_xor(p, 8, 32);
        partial[v] = p;
    }
    if (lo == 0) {
        #pragma unroll
        for (int v = 0; v < 8; ++v) {
            float res = partial[v] + b3v;
            res = fminf(10.0f, fmaxf(-10.0f, res));
            out[i * 256 + uh * 128 + wave * 16 + hi * 8 + v] = res;
        }
    }
}

// ---------------------------------------------------------------------------
extern "C" void kernel_launch(void* const* d_in, const int* in_sizes, int n_in,
                              void* d_out, int out_size, void* d_ws, size_t ws_size,
                              hipStream_t stream) {
    const float* bin  = (const float*)d_in[0];   // (2,512,512) -> (1024,512)
    const float* unit = (const float*)d_in[1];   // (256,512)
    const float* bp_w = (const float*)d_in[2];
    const float* bp_b = (const float*)d_in[3];
    const float* up_w = (const float*)d_in[4];
    const float* up_b = (const float*)d_in[5];
    const float* w1   = (const float*)d_in[6];
    const float* b1   = (const float*)d_in[7];
    const float* w2   = (const float*)d_in[8];
    const float* b2   = (const float*)d_in[9];
    const float* w3   = (const float*)d_in[10];
    const float* b3   = (const float*)d_in[11];
    float* out        = (float*)d_out;

    char* ws = (char*)d_ws;
    float*    Wc  = (float*)(ws + 0);            // 512*256*4 = 524288
    float*    Wd  = (float*)(ws + 524288);       // 524288
    float*    bc  = (float*)(ws + 1048576);      // 1024
    float*    bd  = (float*)(ws + 1049600);      // 1024
    _Float16* Ah  = (_Float16*)(ws + 1050624);   // 1024*256*2 = 524288
    _Float16* Bh  = (_Float16*)(ws + 1574912);   // 256*256*2  = 131072
    _Float16* w2s = (_Float16*)(ws + 1705984);   // 32768*2    = 65536

    fuse_weights_k<<<1153, 256, 0, stream>>>(bp_w, bp_b, up_w, up_b, w1, b1, w2,
                                             Wc, Wd, bc, bd, w2s);
    compute_ab_k<<<1280, 256, 0, stream>>>(bin, unit, Wc, Wd, bc, bd, Ah, Bh);
    pernueron_main_k<<<2048, 256, SMEM_TOTAL, stream>>>(Ah, Bh, w2s, b2, w3, b3, out);
}
